// PosMLP_2774548873284
// MI455X (gfx1250) — matrix operand order repaired and checked by reading, not compiled, not run on hardware
//
#include <hip/hip_runtime.h>

#define H 128
#define TILE 128
#define LSTR 136   // LDS row stride in ushorts (272 B = 17 * 16 B -> bank-conflict-free b128)

typedef __attribute__((ext_vector_type(16))) __bf16 v16bf;
typedef __attribute__((ext_vector_type(8)))  __bf16 v8bf;
typedef __attribute__((ext_vector_type(8)))  float  v8f;

union U4 { uint4 u; v8bf h; };

__device__ __forceinline__ unsigned short f32_to_bf16_bits(float f) {
    unsigned int u = __float_as_uint(f);
    u += 0x7FFFu + ((u >> 16) & 1u);     // round-to-nearest-even
    return (unsigned short)(u >> 16);
}

__device__ __forceinline__ float silu_f(float x) {
    return x / (1.0f + __expf(-x));
}

// Convert pos_w1 and vel_w1 (fp32, [k][n] row-major) to bf16 transposed ([n][k]) in ws.
__global__ void prep_weights_kernel(const float* __restrict__ w1a,
                                    const float* __restrict__ w1b,
                                    unsigned short* __restrict__ ws) {
    int t = threadIdx.x;                  // 256 threads
    #pragma unroll
    for (int i = 0; i < (H * H) / 256; ++i) {
        int id = t + i * 256;
        int k = id >> 7, n = id & (H - 1);
        ws[n * H + k]         = f32_to_bf16_bits(w1a[id]);
        ws[H * H + n * H + k] = f32_to_bf16_bits(w1b[id]);
    }
}

// Fused: rows @ w1 (bf16 WMMA, f32 acc, bias) -> silu -> dot(w2) + b2 -> epilogue.
// EDGE=true : pos_msg scatter-add of (loc[send]-loc[rec])*pm into out[rec].
// EDGE=false: out[n] = loc[n] + vm*vel[n].
template <bool EDGE>
__global__ __launch_bounds__(256)
void mlp_msg_kernel(const float* __restrict__ Ag, int nrows,
                    const unsigned short* __restrict__ w1T,
                    const float* __restrict__ b1,
                    const float* __restrict__ w2,
                    const float* __restrict__ b2,
                    const int* __restrict__ send,
                    const int* __restrict__ rec,
                    const float* __restrict__ loc,
                    const float* __restrict__ vel,
                    float* __restrict__ out) {
    __shared__ unsigned short sA[TILE * LSTR];
    __shared__ unsigned short sB[H * LSTR];
    __shared__ float sMsg[TILE];

    const int t = threadIdx.x;
    const int rowBase = blockIdx.x * TILE;

    // ---- Stage A tile: 128x128 fp32 -> bf16 in LDS (coalesced float4 reads) ----
    #pragma unroll
    for (int i = 0; i < 16; ++i) {
        int c = t + i * 256;          // float4 chunk id, 4096 total
        int row = c >> 5;             // 32 float4 per row
        int col4 = c & 31;
        int grow = rowBase + row;
        if (!EDGE) grow = (grow < nrows) ? grow : (nrows - 1);
        float4 v = ((const float4*)(Ag + (size_t)grow * H))[col4];
        unsigned int p0 = (unsigned int)f32_to_bf16_bits(v.x) |
                          ((unsigned int)f32_to_bf16_bits(v.y) << 16);
        unsigned int p1 = (unsigned int)f32_to_bf16_bits(v.z) |
                          ((unsigned int)f32_to_bf16_bits(v.w) << 16);
        *(uint2*)&sA[row * LSTR + col4 * 4] = make_uint2(p0, p1);
    }

    // ---- Stage B: bf16 w1^T from workspace (L2-resident), padded rows ----
    #pragma unroll
    for (int i = 0; i < 8; ++i) {
        int c = t + i * 256;          // 8-ushort chunk id, 2048 total
        int row = c >> 4;             // 16 chunks per row
        int col8 = c & 15;
        uint4 v = ((const uint4*)(w1T + row * H))[col8];
        *(uint4*)&sB[row * LSTR + col8 * 8] = v;
    }
    __syncthreads();

    const int wave = t >> 5;          // 0..7 -> M-tile (16 rows each)
    const int lane = t & 31;
    const int l15  = lane & 15;
    const int grp  = lane >> 4;       // lane group 0/1

    // A fragments (16x32 bf16, ISA layout): row m = 16*wave + l15
    //   elems 0..7 : k = kk*32 + 8*grp + i ; elems 8..15 : k = kk*32 + 16 + 8*grp + i
    v16bf afrag[4];
    {
        const unsigned short* arow = &sA[(16 * wave + l15) * LSTR];
        #pragma unroll
        for (int kk = 0; kk < 4; ++kk) {
            U4 lo, hi;
            lo.u = *(const uint4*)&arow[kk * 32 + 8 * grp];
            hi.u = *(const uint4*)&arow[kk * 32 + 16 + 8 * grp];
            afrag[kk] = __builtin_shufflevector(lo.h, hi.h,
                0, 1, 2, 3, 4, 5, 6, 7, 8, 9, 10, 11, 12, 13, 14, 15);
        }
    }

    float racc[8];
    #pragma unroll
    for (int r = 0; r < 8; ++r) racc[r] = 0.0f;

    #pragma unroll
    for (int nt = 0; nt < 8; ++nt) {
        const int n = nt * 16 + l15;
        const float bias = b1[n];
        v8f c;
        #pragma unroll
        for (int r = 0; r < 8; ++r) c[r] = bias;

        // B fragment (32x16 bf16): col n, elems 0..7 : k = kk*32 + 8*grp + i ; elems 8..15 : k = kk*32 + 16 + 8*grp + i
        const unsigned short* brow = &sB[n * LSTR];
        #pragma unroll
        for (int kk = 0; kk < 4; ++kk) {
            U4 lo, hi;
            lo.u = *(const uint4*)&brow[kk * 32 + 8 * grp];
            hi.u = *(const uint4*)&brow[kk * 32 + 16 + 8 * grp];
            v16bf bfrag = __builtin_shufflevector(lo.h, hi.h,
                0, 1, 2, 3, 4, 5, 6, 7, 8, 9, 10, 11, 12, 13, 14, 15);
            c = __builtin_amdgcn_wmma_f32_16x16x32_bf16(
                    false, afrag[kk], false, bfrag, (short)0, c, false, false);
        }
        const float wn = w2[n];
        #pragma unroll
        for (int r = 0; r < 8; ++r) racc[r] += silu_f(c[r]) * wn;
    }

    // Reduce the per-row dot over the 16 lanes of the group (bits 0..3 only).
    #pragma unroll
    for (int off = 1; off < 16; off <<= 1) {
        #pragma unroll
        for (int r = 0; r < 8; ++r) racc[r] += __shfl_xor(racc[r], off, 32);
    }
    if (l15 == 0) {
        const float bb = b2[0];
        #pragma unroll
        for (int r = 0; r < 8; ++r)
            sMsg[16 * wave + 8 * grp + r] = racc[r] + bb;   // row = D-layout row
    }
    __syncthreads();

    // ---- Epilogue ----
    if (t < TILE) {
        int idx = rowBase + t;
        if (idx < nrows) {
            float pm = sMsg[t];
            if (EDGE) {
                int s = send[idx], rc = rec[idx];
                #pragma unroll
                for (int c2 = 0; c2 < 3; ++c2) {
                    float d = loc[3 * s + c2] - loc[3 * rc + c2];
                    atomicAdd(&out[3 * rc + c2], d * pm);
                }
            } else {
                #pragma unroll
                for (int c2 = 0; c2 < 3; ++c2)
                    out[3 * idx + c2] = loc[3 * idx + c2] + pm * vel[3 * idx + c2];
            }
        }
    }
}

extern "C" void kernel_launch(void* const* d_in, const int* in_sizes, int n_in,
                              void* d_out, int out_size, void* d_ws, size_t ws_size,
                              hipStream_t stream) {
    const float* x      = (const float*)d_in[0];
    const float* vel    = (const float*)d_in[1];
    const float* loc    = (const float*)d_in[2];
    const float* mes    = (const float*)d_in[3];
    const int*   send   = (const int*)d_in[4];
    const int*   rec    = (const int*)d_in[5];
    const float* pos_w1 = (const float*)d_in[6];
    const float* pos_b1 = (const float*)d_in[7];
    const float* pos_w2 = (const float*)d_in[8];
    const float* pos_b2 = (const float*)d_in[9];
    const float* vel_w1 = (const float*)d_in[10];
    const float* vel_b1 = (const float*)d_in[11];
    const float* vel_w2 = (const float*)d_in[12];
    const float* vel_b2 = (const float*)d_in[13];

    const int Nn = in_sizes[0] / H;   // 100000 nodes
    const int Ee = in_sizes[3] / H;   // 1600000 edges

    unsigned short* wsPos = (unsigned short*)d_ws;   // [H*H] bf16 pos_w1^T
    unsigned short* wsVel = wsPos + H * H;           // [H*H] bf16 vel_w1^T

    prep_weights_kernel<<<1, 256, 0, stream>>>(pos_w1, vel_w1, wsPos);

    // Node kernel first: initializes every out element (loc + vm*vel)...
    mlp_msg_kernel<false><<<(Nn + TILE - 1) / TILE, 256, 0, stream>>>(
        x, Nn, wsVel, vel_b1, vel_w2, vel_b2,
        nullptr, nullptr, loc, vel, (float*)d_out);

    // ...then edge kernel scatter-adds aggregated messages on top.
    mlp_msg_kernel<true><<<(Ee + TILE - 1) / TILE, 256, 0, stream>>>(
        mes, Ee, wsPos, pos_b1, pos_w2, pos_b2,
        send, rec, loc, vel, (float*)d_out);
}
